// AnomalyTransformer_34875134443772
// MI455X (gfx1250) — matrix operand order repaired and checked by reading, not compiled, not run on hardware
//
#include <hip/hip_runtime.h>
#include <math.h>
#include <stdint.h>

typedef __attribute__((ext_vector_type(16))) _Float16 v16h;
typedef __attribute__((ext_vector_type(8)))  _Float16 v8h;
typedef __attribute__((ext_vector_type(4)))  _Float16 v4h;
typedef __attribute__((ext_vector_type(8)))  float    v8f;

// D = A(16x32 f16) * B(32x16 f16) + C(16x16 f32)
#define WMMA16(a, b, c) __builtin_amdgcn_wmma_f32_16x16x32_f16( \
    false, (a), false, (b), (short)0, (c), false, false)

// ---- WMMA fragment loaders (contiguous K order, same for A and B) ----

// A-matrix 16x32 f16: lane L holds row m0+(L&15); K = (L>>4)*16 + e,
// 16 contiguous K (matches load_frag_b).
__device__ __forceinline__ v16h load_frag_a(const _Float16* base, int ld, int m0, int koff) {
    int lane = threadIdx.x & 31;
    const _Float16* p = base + (size_t)(m0 + (lane & 15)) * ld + koff + ((lane >> 4) << 4);
    v16h a;
#pragma unroll
    for (int e = 0; e < 8; ++e) { a[e] = p[e]; a[e + 8] = p[e + 8]; }
    return a;
}

// B-matrix 32x16 f16 from a [N][K] buffer: lane L holds col n0+(L&15);
// K = (L>>4)*16 + e -> 16 contiguous f16 = one 32-byte load.
__device__ __forceinline__ v16h load_frag_b(const _Float16* base, int ldk, int n0, int koff) {
    int lane = threadIdx.x & 31;
    return *(const v16h*)(base + (size_t)(n0 + (lane & 15)) * ldk + koff + ((lane >> 4) << 4));
}

// async 16B global -> LDS copy (CDNA5, tracked by ASYNCcnt)
__device__ __forceinline__ void async_copy_b128(unsigned lds_addr, const void* gaddr) {
    asm volatile("global_load_async_to_lds_b128 %0, %1, off"
                 :: "v"(lds_addr), "v"((unsigned long long)(uintptr_t)gaddr)
                 : "memory");
}
__device__ __forceinline__ void wait_async0() {
    asm volatile("s_wait_asynccnt 0x0" ::: "memory");
}

// ---------------- one-time weight transpose + f32->f16 ----------------
// W[K][N] (f32) -> WT[N][K] (f16). 32x32 tiles via LDS.
__global__ __launch_bounds__(256) void transpose_cvt_f16(
    const float* __restrict__ W, _Float16* __restrict__ WT, int K, int N)
{
    __shared__ float tile[32][33];
    const int n0 = blockIdx.x * 32, k0 = blockIdx.y * 32;
    const int tid = threadIdx.x;
    const int r = tid >> 3, c4 = (tid & 7) * 4;
#pragma unroll
    for (int i = 0; i < 4; ++i)
        tile[r][c4 + i] = W[(size_t)(k0 + r) * N + n0 + c4 + i];
    __syncthreads();
    v4h o;
#pragma unroll
    for (int i = 0; i < 4; ++i) o[i] = (_Float16)tile[c4 + i][r];
    *(v4h*)(WT + (size_t)(n0 + r) * K + k0 + c4) = o;
}

// ---------------- elementwise f32 -> f16 (x16 for layer 0) ----------------
__global__ __launch_bounds__(256) void cvt_f16_kernel(
    const float* __restrict__ in, _Float16* __restrict__ out, int n4)
{
    int idx = blockIdx.x * 256 + threadIdx.x;
    if (idx < n4) {
        float4 v = ((const float4*)in)[idx];
        v4h o;
        o[0] = (_Float16)v.x; o[1] = (_Float16)v.y;
        o[2] = (_Float16)v.z; o[3] = (_Float16)v.w;
        ((v4h*)out)[idx] = o;
    }
}

// ---------------- GEMM: out = act(A16[M,K] * WT[N,K]^T + bias) ----------------
// block tile 64(M) x 128(N), 8 waves as 2x4, wave tile 32x32, K step 32.
// A tile double-buffered via async global->LDS (ASYNCcnt); one barrier/chunk.
// C (f32, row-major) and CT16 (f16, transposed [N][M]) outputs are optional.
__global__ __launch_bounds__(256) void gemm_f16_wmma(
    const _Float16* __restrict__ A16, const _Float16* __restrict__ WT,
    const float* __restrict__ bias, float* __restrict__ C,
    _Float16* __restrict__ CT16,
    int M, int K, int Nout, int relu)
{
    __shared__ _Float16 As[2][64 * 32];  // double-buffered A tile (2 x 4 KB)

    const int tid  = threadIdx.x;
    const int wave = tid >> 5;
    const int wm   = wave >> 2;          // 0..1
    const int wn   = wave & 3;           // 0..3
    const int row_block = blockIdx.x * 64;
    const int col_block = blockIdx.y * 128;

    v8f acc[2][2] = {};

    const int ar = tid >> 2, ac = (tid & 3) * 8;     // A stage: 8 f16 = 16B/thread
    const _Float16* ag = A16 + (size_t)(row_block + ar) * K + ac;
    const unsigned lds0 = (unsigned)(uintptr_t)(&As[0][ar * 32 + ac]);
    const unsigned lds1 = (unsigned)(uintptr_t)(&As[1][ar * 32 + ac]);

    // prologue: chunk 0 -> buf 0
    async_copy_b128(lds0, ag);

    const int nch = K >> 5;
    for (int i = 0; i < nch; ++i) {
        wait_async0();                   // own async write(s) to buf[i&1] done
        __syncthreads();                 // all waves' buf[i&1] visible; prior reads of buf[(i+1)&1] retired
        if (i + 1 < nch)
            async_copy_b128((i + 1) & 1 ? lds1 : lds0, ag + (size_t)(i + 1) * 32);

        const _Float16* buf = (i & 1) ? &As[1][0] : &As[0][0];
        const int k0 = i * 32;
        v16h a0 = load_frag_a(buf, 32, wm * 32,      0);
        v16h a1 = load_frag_a(buf, 32, wm * 32 + 16, 0);
        v16h b0 = load_frag_b(WT, K, col_block + wn * 32,      k0);
        v16h b1 = load_frag_b(WT, K, col_block + wn * 32 + 16, k0);
        acc[0][0] = WMMA16(a0, b0, acc[0][0]);
        acc[0][1] = WMMA16(a0, b1, acc[0][1]);
        acc[1][0] = WMMA16(a1, b0, acc[1][0]);
        acc[1][1] = WMMA16(a1, b1, acc[1][1]);
    }

    // epilogue: C/D layout -> row = (lane>>4)*8 + r, col = lane&15
    const int lane = tid & 31;
#pragma unroll
    for (int mt = 0; mt < 2; ++mt) {
#pragma unroll
        for (int nt = 0; nt < 2; ++nt) {
            int col  = col_block + wn * 32 + nt * 16 + (lane & 15);
            int row0 = row_block + wm * 32 + mt * 16 + ((lane >> 4) << 3);
            float bv = bias ? bias[col] : 0.0f;
            if (C) {
#pragma unroll
                for (int r = 0; r < 8; ++r) {
                    float v = acc[mt][nt][r] + bv;
                    if (relu) v = fmaxf(v, 0.0f);
                    C[(size_t)(row0 + r) * Nout + col] = v;
                }
            }
            if (CT16) {                  // transposed f16: 8 rows of one col -> 16B store
                v8h o;
#pragma unroll
                for (int r = 0; r < 8; ++r) {
                    float v = acc[mt][nt][r] + bv;
                    if (relu) v = fmaxf(v, 0.0f);
                    o[r] = (_Float16)v;
                }
                *(v8h*)(CT16 + (size_t)col * M + row0) = o;
            }
        }
    }
}

// -------- Flash attention: Z = softmax(QK^T/sqrt(dm)) V, per (batch, 64 q-rows) --------
// Q,K: f32 [B*N, dm]; VT: f16 transposed [dm][B*N]; Z: f32 [B*N, dm].
__global__ __launch_bounds__(256, 1) void attn_flash_wmma(
    const float* __restrict__ Q, const float* __restrict__ K,
    const _Float16* __restrict__ VT, float* __restrict__ Z,
    int Nn, int dm, int MN)
{
    __shared__ _Float16 Qs[64 * 64];   // q rows x dm-chunk
    __shared__ _Float16 Ks[64 * 64];   // keys  x dm-chunk
    __shared__ float    Ssc[64 * 64];  // raw scores
    __shared__ _Float16 Ps[64 * 64];   // exp probs (f16) for P*V
    __shared__ float    red[64 * 4];
    __shared__ float    rowm[64], rowl[64], rowa[64], rowt[64];

    const int tid  = threadIdx.x;
    const int lane = tid & 31;
    const int wave = tid >> 5;          // 0..7, owns dm cols [wave*64, wave*64+64)
    const int b    = blockIdx.y;
    const int q0   = blockIdx.x * 64;

    const float* Qb = Q + (size_t)b * Nn * dm;
    const float* Kb = K + (size_t)b * Nn * dm;
    float*       Zb = Z + (size_t)b * Nn * dm;

    if (tid < 64) { rowm[tid] = -3.0e38f; rowl[tid] = 0.0f; }
    v8f zacc[4][4] = {};                // 64 rows x 64 cols per wave
    const float scale = rsqrtf((float)dm);
    __syncthreads();

    for (int kt = 0; kt < Nn; kt += 64) {
        // ---- scores S = Q * K^T over dm in 64-chunks ----
        const int wr = wave >> 1;       // 0..3 : 16-row tile
        const int wc = wave & 1;        // 0..1 : 32-col half
        v8f sacc[2] = {};
        for (int c = 0; c < dm; c += 64) {
            int r = tid >> 2, c16 = (tid & 3) * 16;
            const float* gq = Qb + (size_t)(q0 + r) * dm + c + c16;
            const float* gk = Kb + (size_t)(kt + r) * dm + c + c16;
            _Float16* sq = Qs + r * 64 + c16;
            _Float16* sk = Ks + r * 64 + c16;
#pragma unroll
            for (int i = 0; i < 16; ++i) { sq[i] = (_Float16)gq[i]; sk[i] = (_Float16)gk[i]; }
            __syncthreads();
#pragma unroll
            for (int s = 0; s < 2; ++s) {
                v16h aq  = load_frag_a(Qs, 64, wr * 16, s * 32);
                v16h bk0 = load_frag_b(Ks, 64, wc * 32,      s * 32);
                v16h bk1 = load_frag_b(Ks, 64, wc * 32 + 16, s * 32);
                sacc[0] = WMMA16(aq, bk0, sacc[0]);
                sacc[1] = WMMA16(aq, bk1, sacc[1]);
            }
            __syncthreads();
        }
        // spill scaled scores to LDS
#pragma unroll
        for (int nt = 0; nt < 2; ++nt) {
            int ccol = wc * 32 + nt * 16 + (lane & 15);
            int rrow = wr * 16 + ((lane >> 4) << 3);
#pragma unroll
            for (int r = 0; r < 8; ++r) Ssc[(rrow + r) * 64 + ccol] = sacc[nt][r] * scale;
        }
        __syncthreads();

        // ---- online softmax: 4 threads per row ----
        {
            int row = tid >> 2, qd = tid & 3;
            const float* srow = Ssc + row * 64 + qd * 16;
            float tmax = -3.0e38f;
#pragma unroll
            for (int i = 0; i < 16; ++i) tmax = fmaxf(tmax, srow[i]);
            red[row * 4 + qd] = tmax;
            __syncthreads();
            if (qd == 0) {
                float rmax = fmaxf(fmaxf(red[row * 4], red[row * 4 + 1]),
                                   fmaxf(red[row * 4 + 2], red[row * 4 + 3]));
                float mold = rowm[row];
                float mnew = fmaxf(mold, rmax);
                rowt[row] = mnew;
                rowa[row] = __expf(mold - mnew);
                rowm[row] = mnew;
            }
            __syncthreads();
            float mnew = rowt[row];
            float psum = 0.0f;
            _Float16* prow = Ps + row * 64 + qd * 16;
            v8h pk0, pk1;
#pragma unroll
            for (int i = 0; i < 8; ++i) {
                float p = __expf(srow[i] - mnew);
                pk0[i] = (_Float16)p; psum += p;
            }
#pragma unroll
            for (int i = 0; i < 8; ++i) {
                float p = __expf(srow[i + 8] - mnew);
                pk1[i] = (_Float16)p; psum += p;
            }
            *(v8h*)prow       = pk0;
            *(v8h*)(prow + 8) = pk1;
            red[row * 4 + qd] = psum;
            __syncthreads();
            if (qd == 0)
                rowl[row] = rowl[row] * rowa[row] + red[row * 4] + red[row * 4 + 1]
                          + red[row * 4 + 2] + red[row * 4 + 3];
            __syncthreads();
        }

        // rescale running Z by alpha
#pragma unroll
        for (int rt = 0; rt < 4; ++rt) {
            int rrow = rt * 16 + ((lane >> 4) << 3);
#pragma unroll
            for (int r = 0; r < 8; ++r) {
                float al = rowa[rrow + r];
#pragma unroll
                for (int ct = 0; ct < 4; ++ct) zacc[rt][ct][r] *= al;
            }
        }

        // ---- Z += P(f16) * V : B fragments straight from transposed f16 V ----
#pragma unroll
        for (int s = 0; s < 2; ++s) {
            v16h bv[4];
#pragma unroll
            for (int ct = 0; ct < 4; ++ct) {
                int n  = wave * 64 + ct * 16 + (lane & 15);
                int kb = kt + s * 32 + ((lane >> 4) << 4);
                bv[ct] = *(const v16h*)(VT + (size_t)n * MN + (size_t)b * Nn + kb);
            }
#pragma unroll
            for (int rt = 0; rt < 4; ++rt) {
                v16h ap = load_frag_a(Ps, 64, rt * 16, s * 32);
#pragma unroll
                for (int ct = 0; ct < 4; ++ct) zacc[rt][ct] = WMMA16(ap, bv[ct], zacc[rt][ct]);
            }
        }
        __syncthreads();
    }

    // epilogue: Z /= l
#pragma unroll
    for (int rt = 0; rt < 4; ++rt) {
#pragma unroll
        for (int ct = 0; ct < 4; ++ct) {
            int col   = wave * 64 + ct * 16 + (lane & 15);
            int lrow0 = rt * 16 + ((lane >> 4) << 3);
#pragma unroll
            for (int r = 0; r < 8; ++r) {
                int lr = lrow0 + r;
                Zb[(size_t)(q0 + lr) * dm + col] = zacc[rt][ct][r] / rowl[lr];
            }
        }
    }
}

// ------- Fused residual + LayerNorm, one block per row (n=512); optional f16 copy -------
__global__ __launch_bounds__(256) void ln_kernel(
    const float* __restrict__ X, const float* __restrict__ R,
    const float* __restrict__ g, const float* __restrict__ bta,
    float* __restrict__ out, _Float16* __restrict__ out16, int n)
{
    const int row = blockIdx.x, tid = threadIdx.x;
    const float* xp = X + (size_t)row * n;
    float v0 = xp[tid], v1 = xp[tid + 256];
    if (R) { const float* rp = R + (size_t)row * n; v0 += rp[tid]; v1 += rp[tid + 256]; }

    __shared__ float sb[256];
    __shared__ float stat;
    sb[tid] = v0 + v1;
    __syncthreads();
    for (int off = 128; off > 0; off >>= 1) {
        if (tid < off) sb[tid] += sb[tid + off];
        __syncthreads();
    }
    if (tid == 0) stat = sb[0] / (float)n;
    __syncthreads();
    float m = stat;
    float d0 = v0 - m, d1 = v1 - m;
    sb[tid] = d0 * d0 + d1 * d1;
    __syncthreads();
    for (int off = 128; off > 0; off >>= 1) {
        if (tid < off) sb[tid] += sb[tid + off];
        __syncthreads();
    }
    if (tid == 0) stat = sb[0] / (float)n;
    __syncthreads();
    float inv = rsqrtf(stat + 1e-5f);
    float y0 = d0 * inv * g[tid]       + bta[tid];
    float y1 = d1 * inv * g[tid + 256] + bta[tid + 256];
    out[(size_t)row * n + tid]       = y0;
    out[(size_t)row * n + tid + 256] = y1;
    if (out16) {
        out16[(size_t)row * n + tid]       = (_Float16)y0;
        out16[(size_t)row * n + tid + 256] = (_Float16)y1;
    }
}

// ---------------- Orchestration ----------------
extern "C" void kernel_launch(void* const* d_in, const int* in_sizes, int n_in,
                              void* d_out, int out_size, void* d_ws, size_t ws_size,
                              hipStream_t stream) {
    (void)in_sizes; (void)n_in; (void)out_size; (void)ws_size;
    const int B = 16, N = 1024, d0 = 64, dm = 512, L = 3;
    const int M = B * N;
    const size_t SZ = (size_t)M * dm;

    const float* x   = (const float*)d_in[0];
    const float* Wq0 = (const float*)d_in[1];
    const float* Wk0 = (const float*)d_in[2];
    const float* Wv0 = (const float*)d_in[3];
    // d_in[4] (Ws0) and d_in[8] (Wss): sigma/Gaussian-prior weights — dead code
    const float* Wqs = (const float*)d_in[5];
    const float* Wks = (const float*)d_in[6];
    const float* Wvs = (const float*)d_in[7];
    const float* g1  = (const float*)d_in[9];
    const float* b1  = (const float*)d_in[10];
    const float* Wf  = (const float*)d_in[11];
    const float* bf  = (const float*)d_in[12];
    const float* g2  = (const float*)d_in[13];
    const float* b2  = (const float*)d_in[14];

    float* ws = (float*)d_ws;
    float* h  = ws;            // layer output (f32), next layer's residual
    float* qb = ws + 1 * SZ;
    float* kb = ws + 2 * SZ;
    float* zb = ws + 3 * SZ;   // attention out
    float* z1 = ws + 4 * SZ;   // LN1 out (f32)
    float* fb = ws + 5 * SZ;   // FF out
    _Float16* h16  = (_Float16*)(ws + 6 * SZ);   // f16 layer input  [M, K]
    _Float16* z116 = h16  + SZ;                  // f16 LN1 out      [M, dm]
    _Float16* vt16 = z116 + SZ;                  // f16 V transposed [dm, M]
    _Float16* wq16 = vt16 + SZ;                  // f16 W^T buffers  [dm, K]
    _Float16* wk16 = wq16 + (size_t)dm * dm;
    _Float16* wv16 = wk16 + (size_t)dm * dm;
    _Float16* wf16 = wv16 + (size_t)dm * dm;

    dim3 blk(256);
    dim3 gGemm(M / 64, dm / 128);
    dim3 gAttn(N / 64, B);
    dim3 gLn(M);

    // layer-0 input -> f16
    cvt_f16_kernel<<<dim3((M * d0 / 4 + 255) / 256), blk, 0, stream>>>(x, h16, M * d0 / 4);

    for (int l = 0; l < L; ++l) {
        const float* in = (l == 0) ? x : h;   // f32 residual source
        const int Kdim  = (l == 0) ? d0 : dm;
        const float* Wq = (l == 0) ? Wq0 : (Wqs + (size_t)(l - 1) * dm * dm);
        const float* Wk = (l == 0) ? Wk0 : (Wks + (size_t)(l - 1) * dm * dm);
        const float* Wv = (l == 0) ? Wv0 : (Wvs + (size_t)(l - 1) * dm * dm);

        dim3 gT(dm / 32, Kdim / 32);
        transpose_cvt_f16<<<gT, blk, 0, stream>>>(Wq, wq16, Kdim, dm);
        transpose_cvt_f16<<<gT, blk, 0, stream>>>(Wk, wk16, Kdim, dm);
        transpose_cvt_f16<<<gT, blk, 0, stream>>>(Wv, wv16, Kdim, dm);
        dim3 gTf(dm / 32, dm / 32);
        transpose_cvt_f16<<<gTf, blk, 0, stream>>>(Wf + (size_t)l * dm * dm, wf16, dm, dm);

        gemm_f16_wmma<<<gGemm, blk, 0, stream>>>(h16, wq16, nullptr, qb, nullptr, M, Kdim, dm, 0);
        gemm_f16_wmma<<<gGemm, blk, 0, stream>>>(h16, wk16, nullptr, kb, nullptr, M, Kdim, dm, 0);
        gemm_f16_wmma<<<gGemm, blk, 0, stream>>>(h16, wv16, nullptr, nullptr, vt16, M, Kdim, dm, 0);

        attn_flash_wmma<<<gAttn, blk, 0, stream>>>(qb, kb, vt16, zb, N, dm, M);

        // LN1: residual only when d_in == dm (layers >= 1)
        ln_kernel<<<gLn, blk, 0, stream>>>(zb, (l == 0) ? nullptr : in,
                                           g1 + (size_t)l * dm, b1 + (size_t)l * dm,
                                           z1, z116, dm);

        gemm_f16_wmma<<<gGemm, blk, 0, stream>>>(z116, wf16, bf + (size_t)l * dm,
                                                 fb, nullptr, M, dm, dm, 1);

        float* outp = (l == L - 1) ? (float*)d_out : h;
        ln_kernel<<<gLn, blk, 0, stream>>>(fb, z1,
                                           g2 + (size_t)l * dm, b2 + (size_t)l * dm,
                                           outp, (l == L - 1) ? nullptr : h16, dm);
    }
}
